// MolCrystalScalarGNN_49864570306578
// MI455X (gfx1250) — matrix operand order repaired and checked by reading, not compiled, not run on hardware
//
#include <hip/hip_runtime.h>
#include <hip/hip_bf16.h>

typedef __bf16 bf16;
typedef __attribute__((ext_vector_type(16))) __bf16 v16bf;
typedef __attribute__((ext_vector_type(8)))  __bf16 v8bf;
typedef __attribute__((ext_vector_type(8)))  float  v8f;

#define DIM  128
#define WPAD 136   // padded LDS row (elements) for 128-wide tiles -> conflict-free strided reads
#define EPAD 40    // padded LDS row for 32-wide (rbf/Wedge) tiles
#define GTPB 4     // 32-row tiles per gemm block (amortize sW staging)
#define ETPB 8     // 16-edge tiles per edge block (amortize Wedge staging)

__device__ __forceinline__ float fast_tanh(float x) {
#if __has_builtin(__builtin_amdgcn_tanhf)
    return __builtin_amdgcn_tanhf(x);            // hw v_tanh_f32
#else
    float y;
    asm volatile("v_tanh_f32 %0, %1\n\tv_nop\n\tv_nop" : "=v"(y) : "v"(x));
    return y;
#endif
}

__device__ __forceinline__ float gelu_f(float x) {
    // jax.nn.gelu default (approximate=True, tanh form)
    const float c0 = 0.7978845608028654f;   // sqrt(2/pi)
    const float c1 = 0.044715f;
    return 0.5f * x * (1.0f + fast_tanh(c0 * (x + c1 * x * x * x)));
}

__device__ __forceinline__ v16bf cat16(v8bf lo, v8bf hi) {
    return __builtin_shufflevector(lo, hi, 0,1,2,3,4,5,6,7,8,9,10,11,12,13,14,15);
}

// ---------------------------------------------------------------- utilities
__global__ __launch_bounds__(256) void k_f32_to_bf16(const float* __restrict__ in,
                                                     bf16* __restrict__ out, int n) {
    int i = blockIdx.x * blockDim.x + threadIdx.x;
    if (i < n) out[i] = (bf16)in[i];
}

__global__ __launch_bounds__(256) void k_zero(float* __restrict__ p, int n) {
    int i = blockIdx.x * blockDim.x + threadIdx.x;
    if (i < n) p[i] = 0.0f;
}

// ------------------------------------------------------------- embedding
__global__ __launch_bounds__(256) void k_embed(const int* __restrict__ z, const float* __restrict__ feats,
                        const float* __restrict__ table, const float* __restrict__ W,
                        const float* __restrict__ b, float* __restrict__ x, int N) {
    int t = blockIdx.x * blockDim.x + threadIdx.x;
    int i = t >> 7, j = t & 127;
    if (i >= N) return;
    const float* e = table + (unsigned)z[i] * 5u;
    const float* f = feats + (unsigned)i * 7u;
    float acc = b[j];
#pragma unroll
    for (int k = 0; k < 5; k++) acc += e[k] * W[k * DIM + j];
#pragma unroll
    for (int k = 0; k < 7; k++) acc += f[k] * W[(5 + k) * DIM + j];
    x[(unsigned)i * (unsigned)DIM + j] = acc;
}

// ------------------------------------------------------------- Bessel RBF
__global__ __launch_bounds__(256) void k_rbf(const float* __restrict__ pos, const int* __restrict__ esrc,
                      const int* __restrict__ edst, bf16* __restrict__ rbf, int E) {
    int e = blockIdx.x * blockDim.x + threadIdx.x;
    if (e >= E) return;
    unsigned s = (unsigned)esrc[e] * 3u, t = (unsigned)edst[e] * 3u;
    float dx = pos[s + 0] - pos[t + 0];
    float dy = pos[s + 1] - pos[t + 1];
    float dz = pos[s + 2] - pos[t + 2];
    float dist = sqrtf(dx * dx + dy * dy + dz * dz);
    float d = dist * 0.2f;                       // / CUTOFF
    float d2 = d * d, d4 = d2 * d2, d5 = d4 * d, d6 = d5 * d, d7 = d6 * d;
    // p=6: a=-28, b=48, c=-21
    float env = (1.0f / d - 28.0f * d5 + 48.0f * d6 - 21.0f * d7) * (d < 1.0f ? 1.0f : 0.0f);
    const float PI = 3.14159265358979323846f;
    bf16* o = rbf + (unsigned)e * 32u;
#pragma unroll
    for (int r = 0; r < 32; r++) {
        o[r] = (bf16)(env * __sinf((float)(r + 1) * PI * d));   // hw v_sin_f32
    }
}

// --------------------------------------------------- WMMA GEMM (K = N = 128)
// Y = epi(A[rows,128] @ W[128,128]); epi: +bias, gelu, +res (compile-time).
// 256 threads = 8 waves; wave w -> cols [16w,16w+16). Block stages W once,
// then loops over GTPB row-tiles of 32 rows. Two independent accumulator
// chains per wave share the B fragment (no WMMA->WMMA RAW NOPs).
template <bool BIAS, bool ACT, bool RES>
__global__ __launch_bounds__(256) void k_gemm128(const float* __restrict__ A, const bf16* __restrict__ Wb,
                          const float* __restrict__ bias, const float* __restrict__ res,
                          float* __restrict__ Y, int rows) {
    __shared__ bf16 sA[32 * WPAD];        // A tile, bf16
    __shared__ bf16 sW[128 * WPAD];       // W transposed: sW[n*WPAD + k]
    int tid = threadIdx.x;

    for (int idx = tid; idx < 128 * 128; idx += 256) {
        int k = idx >> 7, nn = idx & 127;
        sW[nn * WPAD + k] = Wb[idx];
    }

    int wave = tid >> 5, lane = tid & 31;
    int n0 = wave * 16;
    int m = lane & 15, hi = lane >> 4;
    int khalf = hi * 8;
    int n = lane & 15;
    unsigned col = (unsigned)(n0 + n);
    float bc = BIAS ? bias[col] : 0.0f;

    int tEnd = blockIdx.x * GTPB + GTPB;
    for (int tile = blockIdx.x * GTPB; tile < tEnd; ++tile) {
        int m0 = tile * 32;
        if (m0 >= rows) break;

#pragma unroll
        for (int idx = tid; idx < 32 * 128; idx += 256) {
            int r = idx >> 7, c = idx & 127;
            int row = m0 + r; if (row >= rows) row = rows - 1;
            sA[r * WPAD + c] = (bf16)A[(unsigned)row * (unsigned)DIM + (unsigned)c];
        }
        __syncthreads();   // sA ready (and sW on first iteration)

        v8f acc0 = {}, acc1 = {};
#pragma unroll
        for (int k0 = 0; k0 < 128; k0 += 32) {
            int kb = k0 + khalf;
            v16bf b  = cat16(*(const v8bf*)&sW[(n0 + n) * WPAD + kb],
                             *(const v8bf*)&sW[(n0 + n) * WPAD + kb + 16]);
            v16bf a0 = cat16(*(const v8bf*)&sA[m * WPAD + k0 + khalf],
                             *(const v8bf*)&sA[m * WPAD + k0 + 16 + khalf]);
            v16bf a1 = cat16(*(const v8bf*)&sA[(16 + m) * WPAD + k0 + khalf],
                             *(const v8bf*)&sA[(16 + m) * WPAD + k0 + 16 + khalf]);
            acc0 = __builtin_amdgcn_wmma_f32_16x16x32_bf16(false, a0, false, b,
                                                           (short)0, acc0, false, false);
            acc1 = __builtin_amdgcn_wmma_f32_16x16x32_bf16(false, a1, false, b,
                                                           (short)0, acc1, false, false);
        }

        if (m0 + 32 <= rows) {            // uniform fast path (rows % 32 == 0 here)
#pragma unroll
            for (int t = 0; t < 2; t++) {
                unsigned obase = (unsigned)(m0 + t * 16 + hi * 8) * (unsigned)DIM + col;
#pragma unroll
                for (int r = 0; r < 8; r++) {
                    unsigned o = obase + (unsigned)r * (unsigned)DIM;
                    float v = t ? acc1[r] : acc0[r];
                    if (BIAS) v += bc;
                    if (ACT)  v = gelu_f(v);
                    if (RES)  v += res[o];
                    Y[o] = v;
                }
            }
        } else {
#pragma unroll
            for (int t = 0; t < 2; t++) {
                int mbase = m0 + t * 16 + hi * 8;
#pragma unroll
                for (int r = 0; r < 8; r++) {
                    int row = mbase + r;
                    if (row >= rows) continue;
                    unsigned o = (unsigned)row * (unsigned)DIM + col;
                    float v = t ? acc1[r] : acc0[r];
                    if (BIAS) v += bc;
                    if (ACT)  v = gelu_f(v);
                    if (RES)  v += res[o];
                    Y[o] = v;
                }
            }
        }
        __syncthreads();   // protect sA before next overwrite
    }
}

// ---------------------------------------- fused edge: filt GEMM + msg + scatter
// Block stages Wedge once, then loops over ETPB tiles of 16 edges:
// edge_filt = rbf[16,32] @ Wedge[32,128] + bedge; msg = gelu(h[src]*filt);
// atomicAdd agg[dst]. Gathers use 32-bit element offsets and are issued in a
// batch before use.
__global__ __launch_bounds__(256) void k_edge(const bf16* __restrict__ rbf, const bf16* __restrict__ Wedge,
                       const float* __restrict__ bedge, const float* __restrict__ h,
                       const int* __restrict__ esrc, const int* __restrict__ edst,
                       float* __restrict__ agg, int E) {
    __shared__ bf16 sW[128 * EPAD];       // Wedge transposed: sW[n*EPAD + k]
    __shared__ bf16 sR[16 * EPAD];        // rbf tile
    __shared__ int  sSrc[16], sDst[16];
    int tid = threadIdx.x;

#pragma unroll
    for (int idx = tid; idx < 32 * 128; idx += 256) {
        int k = idx >> 7, nn = idx & 127;
        sW[nn * EPAD + k] = Wedge[idx];
    }

    int wave = tid >> 5, lane = tid & 31;
    int n0 = wave * 16;
    int m = lane & 15, hi = lane >> 4;
    int khalf = hi * 8;
    int n = lane & 15;
    unsigned col = (unsigned)(n0 + n);
    float bc = bedge[col];

    int tEnd = blockIdx.x * ETPB + ETPB;
    for (int tile = blockIdx.x * ETPB; tile < tEnd; ++tile) {
        int m0 = tile * 16;
        if (m0 >= E) break;

        if (tid < 16) {
            int e = m0 + tid;
            sSrc[tid] = (e < E) ? esrc[e] : 0;
            sDst[tid] = (e < E) ? edst[e] : 0;
        }
#pragma unroll
        for (int idx = tid; idx < 16 * 32; idx += 256) {
            int r = idx >> 5, c = idx & 31;
            int e = m0 + r; if (e >= E) e = E - 1;
            sR[r * EPAD + c] = rbf[(unsigned)e * 32u + (unsigned)c];
        }
        __syncthreads();   // sR/indices ready (and sW on first iteration)

        v16bf a = cat16(*(const v8bf*)&sR[m * EPAD + khalf],
                        *(const v8bf*)&sR[m * EPAD + 16 + khalf]);
        int kb = khalf;
        v16bf b = cat16(*(const v8bf*)&sW[(n0 + n) * EPAD + kb],
                        *(const v8bf*)&sW[(n0 + n) * EPAD + kb + 16]);
        v8f acc = {};
        acc = __builtin_amdgcn_wmma_f32_16x16x32_bf16(false, a, false, b,
                                                      (short)0, acc, false, false);

        unsigned hoff[8], aoff[8];
#pragma unroll
        for (int r = 0; r < 8; r++) {
            hoff[r] = (unsigned)sSrc[hi * 8 + r] * (unsigned)DIM + col;
            aoff[r] = (unsigned)sDst[hi * 8 + r] * (unsigned)DIM + col;
        }

        if (m0 + 16 <= E) {               // uniform fast path (E % 16 == 0 here)
            float hv[8];
#pragma unroll
            for (int r = 0; r < 8; r++) hv[r] = h[hoff[r]];
#pragma unroll
            for (int r = 0; r < 8; r++) {
                float mv = gelu_f(hv[r] * (acc[r] + bc));
                atomicAdd(&agg[aoff[r]], mv);
            }
        } else {
#pragma unroll
            for (int r = 0; r < 8; r++) {
                int em = hi * 8 + r;
                if (m0 + em >= E) continue;
                float mv = gelu_f(h[hoff[r]] * (acc[r] + bc));
                atomicAdd(&agg[aoff[r]], mv);
            }
        }
        __syncthreads();   // protect sR/indices before next overwrite
    }
}

// ------------------------------------------------------------- gathers
__global__ __launch_bounds__(256) void k_gather(const float* __restrict__ in, const int* __restrict__ idx,
                         float* __restrict__ out, int rows) {
    int t = blockIdx.x * blockDim.x + threadIdx.x;
    int r = t >> 7, c = t & 127;
    if (r >= rows) return;
    out[(unsigned)r * (unsigned)DIM + c] = in[(unsigned)idx[r] * (unsigned)DIM + (unsigned)c];
}

__global__ __launch_bounds__(256) void k_periodize(const float* __restrict__ xin, const int* __restrict__ tile,
                            const float* __restrict__ aux, float* __restrict__ x, int N) {
    int t = blockIdx.x * blockDim.x + threadIdx.x;
    int i = t >> 7, c = t & 127;
    if (i >= N) return;
    float v = (c == 127) ? aux[i] : xin[(unsigned)tile[i] * (unsigned)DIM + (unsigned)c];
    x[(unsigned)i * (unsigned)DIM + c] = v;
}

__global__ __launch_bounds__(256) void k_final(const float* __restrict__ xin, const int* __restrict__ tile,
                        const int* __restrict__ inside, const float* __restrict__ aux,
                        float* __restrict__ out, int rows) {
    int t = blockIdx.x * blockDim.x + threadIdx.x;
    int r = t >> 7, c = t & 127;
    if (r >= rows) return;
    int node = inside[r];
    float v = (c == 127) ? aux[node] : xin[(unsigned)tile[node] * (unsigned)DIM + (unsigned)c];
    out[(unsigned)r * (unsigned)DIM + c] = v;
}

// ======================================================================
extern "C" void kernel_launch(void* const* d_in, const int* in_sizes, int n_in,
                              void* d_out, int out_size, void* d_ws, size_t ws_size,
                              hipStream_t stream) {
    const int*   z_types   = (const int*)  d_in[0];
    const float* z_feats   = (const float*)d_in[1];
    const float* pos       = (const float*)d_in[2];
    /* d_in[3] = batch (unused) */
    const float* aux_ind   = (const float*)d_in[4];
    const int*   esrc      = (const int*)  d_in[5];
    const int*   edst      = (const int*)  d_in[6];
    const int*   inside    = (const int*)  d_in[7];
    const int*   tile_map  = (const int*)  d_in[8];
    const float* emb_table = (const float*)d_in[9];
    const float* emb_W     = (const float*)d_in[10];
    const float* emb_b     = (const float*)d_in[11];
    const float* mlp0_W    = (const float*)d_in[12];
    const float* mlp0_b    = (const float*)d_in[13];
    const float* Wnode     = (const float*)d_in[14];
    const float* Wedge     = (const float*)d_in[15];
    const float* bedge     = (const float*)d_in[16];
    const float* Wout      = (const float*)d_in[17];
    const float* fcW       = (const float*)d_in[18];
    const float* fcb       = (const float*)d_in[19];

    const int N   = in_sizes[0];          // 27648 nodes
    const int E   = in_sizes[5];          // 800000 edges
    const int NIN = in_sizes[7];          // 1024 inside nodes
    const int NCONV = 2, L = 2;

    // ---- workspace carve-out
    char* wsb = (char*)d_ws;
    size_t off = 0;
    auto carve = [&](size_t bytes) -> char* {
        char* p = wsb + off;
        off = (off + bytes + 255) & ~(size_t)255;
        return p;
    };
    float* xA   = (float*)carve((size_t)N * DIM * 4);
    float* xB   = (float*)carve((size_t)N * DIM * 4);
    float* hbuf = (float*)carve((size_t)N * DIM * 4);
    float* agg  = (float*)carve((size_t)N * DIM * 4);
    float* xinA = (float*)carve((size_t)NIN * DIM * 4);
    float* xinB = (float*)carve((size_t)NIN * DIM * 4);
    bf16*  rbf  = (bf16*) carve((size_t)E * 32 * 2);
    bf16*  b_mlp0 = (bf16*)carve((size_t)L * DIM * DIM * 2);
    bf16*  b_node = (bf16*)carve((size_t)NCONV * DIM * DIM * 2);
    bf16*  b_edge = (bf16*)carve((size_t)NCONV * 32 * DIM * 2);
    bf16*  b_out  = (bf16*)carve((size_t)NCONV * DIM * DIM * 2);
    bf16*  b_fc   = (bf16*)carve((size_t)NCONV * L * DIM * DIM * 2);
    (void)ws_size; (void)n_in; (void)out_size;

    const int TB = 256;
    auto cdiv = [](int a, int b) { return (a + b - 1) / b; };

    // ---- one-time weight conversion to bf16
    k_f32_to_bf16<<<cdiv(L * DIM * DIM, TB), TB, 0, stream>>>(mlp0_W, b_mlp0, L * DIM * DIM);
    k_f32_to_bf16<<<cdiv(NCONV * DIM * DIM, TB), TB, 0, stream>>>(Wnode, b_node, NCONV * DIM * DIM);
    k_f32_to_bf16<<<cdiv(NCONV * 32 * DIM, TB), TB, 0, stream>>>(Wedge, b_edge, NCONV * 32 * DIM);
    k_f32_to_bf16<<<cdiv(NCONV * DIM * DIM, TB), TB, 0, stream>>>(Wout, b_out, NCONV * DIM * DIM);
    k_f32_to_bf16<<<cdiv(NCONV * L * DIM * DIM, TB), TB, 0, stream>>>(fcW, b_fc, NCONV * L * DIM * DIM);

    // ---- embedding block
    k_embed<<<cdiv(N * DIM, TB), TB, 0, stream>>>(z_types, z_feats, emb_table, emb_W, emb_b, xA, N);

    float* x = xA; float* xalt = xB;
    for (int l = 0; l < L; l++) {          // res_mlp: x = x + gelu(x@W + b)
        k_gemm128<true, true, true><<<cdiv(N, 32 * GTPB), TB, 0, stream>>>(
            x, b_mlp0 + (size_t)l * DIM * DIM, mlp0_b + l * DIM, x, xalt, N);
        float* t = x; x = xalt; xalt = t;
    }

    // ---- radial basis (loop-invariant)
    k_rbf<<<cdiv(E, TB), TB, 0, stream>>>(pos, esrc, edst, rbf, E);

    float* xin = xinA; float* xinalt = xinB;
    for (int n = 0; n < NCONV; n++) {
        // h = x @ Wnode[n]
        k_gemm128<false, false, false><<<cdiv(N, 32 * GTPB), TB, 0, stream>>>(
            x, b_node + (size_t)n * DIM * DIM, nullptr, nullptr, hbuf, N);
        // agg = segment_sum(gelu(h[src] * (rbf@Wedge+bedge)), dst)
        k_zero<<<cdiv(N * DIM, TB), TB, 0, stream>>>(agg, N * DIM);
        k_edge<<<cdiv(E, 16 * ETPB), TB, 0, stream>>>(rbf, b_edge + (size_t)n * 32 * DIM,
                                                      bedge + n * DIM, hbuf, esrc, edst, agg, E);
        // x = x + agg @ Wout[n]
        k_gemm128<false, false, true><<<cdiv(N, 32 * GTPB), TB, 0, stream>>>(
            agg, b_out + (size_t)n * DIM * DIM, nullptr, x, xalt, N);
        { float* t = x; x = xalt; xalt = t; }
        // fc res_mlp on inside nodes
        k_gather<<<cdiv(NIN * DIM, TB), TB, 0, stream>>>(x, inside, xin, NIN);
        for (int l = 0; l < L; l++) {
            k_gemm128<true, true, true><<<cdiv(NIN, 32 * GTPB), TB, 0, stream>>>(
                xin, b_fc + (size_t)(n * L + l) * DIM * DIM, fcb + (n * L + l) * DIM,
                xin, xinalt, NIN);
            float* t = xin; xin = xinalt; xinalt = t;
        }
        if (n < NCONV - 1) {
            k_periodize<<<cdiv(N * DIM, TB), TB, 0, stream>>>(xin, tile_map, aux_ind, x, N);
        } else {
            k_final<<<cdiv(NIN * DIM, TB), TB, 0, stream>>>(xin, tile_map, inside, aux_ind,
                                                            (float*)d_out, NIN);
        }
    }
}
